// QuantumGraphNeuralNetwork_41764261986827
// MI455X (gfx1250) — matrix operand order repaired and checked by reading, not compiled, not run on hardware
//
#include <hip/hip_runtime.h>

// ---------------------------------------------------------------------------
// Quantum GNN edge-MLP on MI455X (gfx1250), wave32 + v_wmma_f32_16x16x32_f16.
// Transposed state: hT [64 features x 16 edges]; each wave processes TWO
// 16-edge tiles so every weight A-fragment (ds_load_b128 x2) feeds 2 WMMAs.
// Activation f32->f16 via v_cvt_pk_rtz_f16_f32 (1 VALU op per 2 elements).
// ---------------------------------------------------------------------------

#define NODE_DIM 128
#define ENT_DIM  64
#define N_LAYERS 3

typedef __attribute__((ext_vector_type(16))) _Float16 v16h;
typedef __attribute__((ext_vector_type(2)))  __fp16   v2fp;   // cvt_pkrtz result type
typedef __attribute__((ext_vector_type(8)))  float    v8f;

union FragU { uint4 u[2]; unsigned w[8]; v16h v; };

// ---- LDS layout (bytes) ----------------------------------------------------
// Weight fragment slot = 1024B: two 512B chunks; chunk h: lane*16 = elems 8h..8h+7
#define OFF_W1  0
#define SZ_W1   (3 * 32 * 1024)              // 3 layers x 8 kt x 4 ft
#define OFF_W2  (OFF_W1 + SZ_W1)             // 98304
#define SZ_W2   (3 * 8 * 1024)               // 3 layers x 2 kt x 4 ft
#define OFF_M   (OFF_W2 + SZ_W2)             // 122880
#define SZ_M    (3 * 8 * 1024)
#define OFF_B1  (OFF_M + SZ_M)               // 147456
#define OFF_B2  (OFF_B1 + 3 * 64 * 4)
#define OFF_G   (OFF_B2 + 3 * 64 * 4)
#define OFF_BE  (OFF_G  + 3 * 64 * 4)
#define OFF_H   (OFF_BE + 3 * 64 * 4)        // 150528
#define H_EDGE_STRIDE 144                    // 72 halfs/edge (pad: bank-conflict-free)
#define H_TILE_BYTES  (16 * H_EDGE_STRIDE)   // 2304
#define H_WAVE_BYTES  (2 * H_TILE_BYTES)     // 4608: two tiles per wave
#define LDS_TOTAL (OFF_H + 8 * H_WAVE_BYTES) // 187392 (< 320KB WGP LDS)

// RNE scalar convert (cold path: one-time weight swizzle)
__device__ __forceinline__ unsigned short f2h(float x) {
  _Float16 h = (_Float16)x;
  return __builtin_bit_cast(unsigned short, h);
}

// One-instruction packed convert: v_cvt_pk_rtz_f16_f32
__device__ __forceinline__ unsigned pk2(float a, float b) {
  v2fp h = __builtin_amdgcn_cvt_pkrtz(a, b);
  return __builtin_bit_cast(unsigned, h);
}

__device__ __forceinline__ uint4 pack8h(v8f v) {
  uint4 r;
  r.x = pk2(v[0], v[1]);
  r.y = pk2(v[2], v[3]);
  r.z = pk2(v[4], v[5]);
  r.w = pk2(v[6], v[7]);
  return r;
}

__device__ __forceinline__ v16h load_afrag(const char* base, int lane) {
  FragU f;
  f.u[0] = *(const uint4*)(base + lane * 16);
  f.u[1] = *(const uint4*)(base + 512 + lane * 16);
  return f.v;
}

__device__ __forceinline__ v16h load_hfrag(const char* p) {
  FragU f;
  f.u[0] = ((const uint4*)p)[0];
  f.u[1] = ((const uint4*)p)[1];
  return f.v;
}

// B-fragment from 16 contiguous f32 in global (gathered node features):
// 4x global_load_b128 + 8x v_cvt_pk_rtz_f16_f32
__device__ __forceinline__ v16h cvt_frag16(const float* p) {
  const float4* p4 = (const float4*)p;
  float4 q0 = p4[0], q1 = p4[1], q2 = p4[2], q3 = p4[3];
  FragU r;
  r.w[0] = pk2(q0.x, q0.y);
  r.w[1] = pk2(q0.z, q0.w);
  r.w[2] = pk2(q1.x, q1.y);
  r.w[3] = pk2(q1.z, q1.w);
  r.w[4] = pk2(q2.x, q2.y);
  r.w[5] = pk2(q2.z, q2.w);
  r.w[6] = pk2(q3.x, q3.y);
  r.w[7] = pk2(q3.z, q3.w);
  return r.v;
}

__device__ __forceinline__ v8f wmma16x16x32(v16h a, v16h b, v8f c) {
  return __builtin_amdgcn_wmma_f32_16x16x32_f16(
      false, a, false, b, (short)0, c, false, false);
}

__device__ __forceinline__ void zero4(v8f (&d)[4]) {
#pragma unroll
  for (int ft = 0; ft < 4; ++ft) d[ft] = (v8f){0, 0, 0, 0, 0, 0, 0, 0};
}

// K=64 GEMM against per-wave LDS activation buffers, both half-tiles share A
__device__ __forceinline__ void gemm64_dual(const char* wbase, const char* hbA,
                                            const char* hbB, int lane, int n, int g,
                                            v8f (&dA)[4], v8f (&dB)[4]) {
#pragma unroll
  for (int kt = 0; kt < 2; ++kt) {
    v16h bA = load_hfrag(hbA + n * H_EDGE_STRIDE + kt * 64 + g * 32);
    v16h bB = load_hfrag(hbB + n * H_EDGE_STRIDE + kt * 64 + g * 32);
#pragma unroll
    for (int ft = 0; ft < 4; ++ft) {
      v16h a = load_afrag(wbase + (kt * 4 + ft) * 1024, lane);
      dA[ft] = wmma16x16x32(a, bA, dA[ft]);
      dB[ft] = wmma16x16x32(a, bB, dB[ft]);
    }
  }
}

__device__ __forceinline__ void bias_tanh_store(v8f (&d)[4], const float* b1l,
                                                char* hb, int n, int g) {
#pragma unroll
  for (int ft = 0; ft < 4; ++ft) {
    const float* bp = b1l + ft * 16 + g * 8;
    float4 b0 = *(const float4*)bp;
    float4 b1v = *(const float4*)(bp + 4);
    float bv[8] = {b0.x, b0.y, b0.z, b0.w, b1v.x, b1v.y, b1v.z, b1v.w};
    v8f h;
#pragma unroll
    for (int r = 0; r < 8; ++r) h[r] = tanhf(d[ft][r] + bv[r]);
    *(uint4*)(hb + n * H_EDGE_STRIDE + ft * 32 + g * 16) = pack8h(h);
  }
}

__device__ __forceinline__ void bias_ln_store(v8f (&d)[4], const float* b2l,
                                              const float* gl, const float* el,
                                              char* hb, int n, int g) {
  float s = 0.f, s2 = 0.f;
#pragma unroll
  for (int ft = 0; ft < 4; ++ft) {
    const float* bp = b2l + ft * 16 + g * 8;
    float4 b0 = *(const float4*)bp;
    float4 b1v = *(const float4*)(bp + 4);
    float bv[8] = {b0.x, b0.y, b0.z, b0.w, b1v.x, b1v.y, b1v.z, b1v.w};
#pragma unroll
    for (int r = 0; r < 8; ++r) {
      float v = d[ft][r] + bv[r];
      d[ft][r] = v;
      s += v; s2 += v * v;
    }
  }
  s  += __shfl_xor(s, 16);
  s2 += __shfl_xor(s2, 16);
  float mu   = s * 0.015625f;
  float var  = s2 * 0.015625f - mu * mu;
  float rstd = rsqrtf(var + 1e-5f);
#pragma unroll
  for (int ft = 0; ft < 4; ++ft) {
    const float* gp = gl + ft * 16 + g * 8;
    const float* ep = el + ft * 16 + g * 8;
    float4 g0 = *(const float4*)gp;  float4 g1 = *(const float4*)(gp + 4);
    float4 e0 = *(const float4*)ep;  float4 e1 = *(const float4*)(ep + 4);
    float gv[8] = {g0.x, g0.y, g0.z, g0.w, g1.x, g1.y, g1.z, g1.w};
    float ev[8] = {e0.x, e0.y, e0.z, e0.w, e1.x, e1.y, e1.z, e1.w};
    v8f h;
#pragma unroll
    for (int r = 0; r < 8; ++r) h[r] = (d[ft][r] - mu) * rstd * gv[r] + ev[r];
    *(uint4*)(hb + n * H_EDGE_STRIDE + ft * 32 + g * 16) = pack8h(h);
  }
}

// Bell projection on features 0..3 (d[0][0..3] of g==0 lanes), then accumulate
__device__ __forceinline__ void bell_accum(v8f (&d)[4], v8f (&tot)[4], int g) {
  float m0 = d[0][0], m1 = d[0][1], m2 = d[0][2], m3 = d[0][3];
  const float is2 = 0.7071067811865476f;
  float p0 = (m0 + m3) * is2, p1 = (m0 - m3) * is2;
  float p2 = (m1 + m2) * is2, p3 = (m1 - m2) * is2;
  bool lo = (g == 0);
  d[0][0] = lo ? p0 : m0;
  d[0][1] = lo ? p1 : m1;
  d[0][2] = lo ? p2 : m2;
  d[0][3] = lo ? p3 : m3;
#pragma unroll
  for (int ft = 0; ft < 4; ++ft) tot[ft] += d[ft];
}

__device__ __forceinline__ void norm_store(v8f (&tot)[4], float* op) {
  float q = 0.f;
#pragma unroll
  for (int ft = 0; ft < 4; ++ft)
#pragma unroll
    for (int r = 0; r < 8; ++r) q += tot[ft][r] * tot[ft][r];
  q += __shfl_xor(q, 16);
  float scale = 1.0f / (sqrtf(q) + 1e-8f);
#pragma unroll
  for (int ft = 0; ft < 4; ++ft) {
    float4 o0, o1;
    o0.x = tot[ft][0] * scale; o0.y = tot[ft][1] * scale;
    o0.z = tot[ft][2] * scale; o0.w = tot[ft][3] * scale;
    o1.x = tot[ft][4] * scale; o1.y = tot[ft][5] * scale;
    o1.z = tot[ft][6] * scale; o1.w = tot[ft][7] * scale;
    *(float4*)(op + ft * 16)     = o0;
    *(float4*)(op + ft * 16 + 4) = o1;
  }
}

__global__ __launch_bounds__(256, 1)
void qgnn_kernel(const float* __restrict__ nf, const int* __restrict__ ei,
                 const float* __restrict__ W1, const float* __restrict__ b1,
                 const float* __restrict__ W2, const float* __restrict__ b2,
                 const float* __restrict__ gm, const float* __restrict__ bt,
                 const float* __restrict__ Mw, float* __restrict__ out,
                 int n_edges, int n_pairs) {
  extern __shared__ char smem[];

  const int tid  = threadIdx.x;
  const int lane = tid & 31;
  const int wave = tid >> 5;
  const int n = lane & 15;   // edge-in-tile (WMMA N column)
  const int g = lane >> 4;   // lane group: feature/K half select

  // ---- One-time per block: swizzle W1^T/W2^T/M^T into LDS A-fragments -----
  for (int idx = tid; idx < 144 * 32; idx += 256) {
    int frag = idx >> 5, ln = idx & 31;
    int nn = ln & 15, gg = ln >> 4;
    const float* src; char* dst; int kt, ft;
    if (frag < 96) {
      int l = frag >> 5, r = frag & 31; kt = r >> 2; ft = r & 3;
      src = W1 + l * (256 * 64);
      dst = smem + OFF_W1 + frag * 1024;
    } else if (frag < 120) {
      int fi = frag - 96; int l = fi >> 3, r = fi & 7; kt = r >> 2; ft = r & 3;
      src = W2 + l * 4096;
      dst = smem + OFF_W2 + fi * 1024;
    } else {
      int fi = frag - 120; int l = fi >> 3, r = fi & 7; kt = r >> 2; ft = r & 3;
      src = Mw + l * 4096;
      dst = smem + OFF_M + fi * 1024;
    }
    int m = ft * 16 + nn;
    int k0 = kt * 32;
    unsigned int w[8];
#pragma unroll
    for (int pr = 0; pr < 8; ++pr) {
      int e0 = 2 * pr, e1 = 2 * pr + 1;
      int ka = k0 + gg * 16 + e0;
      int kb = k0 + gg * 16 + e1;
      w[pr] = (unsigned)f2h(src[ka * 64 + m]) | ((unsigned)f2h(src[kb * 64 + m]) << 16);
    }
    uint4 c0; c0.x = w[0]; c0.y = w[1]; c0.z = w[2]; c0.w = w[3];
    uint4 c1; c1.x = w[4]; c1.y = w[5]; c1.z = w[6]; c1.w = w[7];
    *(uint4*)(dst + ln * 16)       = c0;
    *(uint4*)(dst + 512 + ln * 16) = c1;
  }
  for (int i = tid; i < 3 * 64; i += 256) {
    ((float*)(smem + OFF_B1))[i] = b1[i];
    ((float*)(smem + OFF_B2))[i] = b2[i];
    ((float*)(smem + OFF_G))[i]  = gm[i];
    ((float*)(smem + OFF_BE))[i] = bt[i];
  }
  __syncthreads();

  char* hbufA = smem + OFF_H + wave * H_WAVE_BYTES;
  char* hbufB = hbufA + H_TILE_BYTES;
  const int gwave  = blockIdx.x * 8 + wave;
  const int nwaves = gridDim.x * 8;

  for (int pair = gwave; pair < n_pairs; pair += nwaves) {
    const int eA = pair * 32 + n;
    const int eB = eA + 16;
    const float* prowA = nf + (size_t)ei[eA] * NODE_DIM;
    const float* pcolA = nf + (size_t)ei[n_edges + eA] * NODE_DIM;
    const float* prowB = nf + (size_t)ei[eB] * NODE_DIM;
    const float* pcolB = nf + (size_t)ei[n_edges + eB] * NODE_DIM;

    v8f totA[4], totB[4];
    zero4(totA); zero4(totB);

    for (int l = 0; l < N_LAYERS; ++l) {
      const float* b1l = (const float*)(smem + OFF_B1) + l * 64;
      const float* b2l = (const float*)(smem + OFF_B2) + l * 64;
      const float* gl  = (const float*)(smem + OFF_G)  + l * 64;
      const float* el  = (const float*)(smem + OFF_BE) + l * 64;

      // -------- GEMM1: G1 = W1^T x combined^T  (K = 256), both halves -----
      v8f dA[4], dB[4];
      zero4(dA); zero4(dB);
      const char* w1base = smem + OFF_W1 + l * 32 * 1024;
      for (int kt = 0; kt < 8; ++kt) {
        const int off = (kt < 4) ? kt * 32 + g * 16 : (kt - 4) * 32 + g * 16;
        const float* pA = (kt < 4) ? prowA + off : pcolA + off;
        const float* pB = (kt < 4) ? prowB + off : pcolB + off;
        v16h bA = cvt_frag16(pA);
        v16h bB = cvt_frag16(pB);
#pragma unroll
        for (int ft = 0; ft < 4; ++ft) {
          v16h a = load_afrag(w1base + (kt * 4 + ft) * 1024, lane);
          dA[ft] = wmma16x16x32(a, bA, dA[ft]);
          dB[ft] = wmma16x16x32(a, bB, dB[ft]);
        }
      }
      bias_tanh_store(dA, b1l, hbufA, n, g);
      bias_tanh_store(dB, b1l, hbufB, n, g);

      // -------- GEMM2: G2 = W2^T x h1^T  (K = 64) -------------------------
      zero4(dA); zero4(dB);
      gemm64_dual(smem + OFF_W2 + l * 8 * 1024, hbufA, hbufB, lane, n, g, dA, dB);
      bias_ln_store(dA, b2l, gl, el, hbufA, n, g);
      bias_ln_store(dB, b2l, gl, el, hbufB, n, g);

      // -------- GEMM3: G3 = M^T x h3^T  (K = 64) --------------------------
      zero4(dA); zero4(dB);
      gemm64_dual(smem + OFF_M + l * 8 * 1024, hbufA, hbufB, lane, n, g, dA, dB);
      bell_accum(dA, totA, g);
      bell_accum(dB, totB, g);
    }

    norm_store(totA, out + (size_t)eA * ENT_DIM + g * 8);
    norm_store(totB, out + (size_t)eB * ENT_DIM + g * 8);
  }
}

extern "C" void kernel_launch(void* const* d_in, const int* in_sizes, int n_in,
                              void* d_out, int out_size, void* d_ws, size_t ws_size,
                              hipStream_t stream) {
  (void)n_in; (void)out_size; (void)d_ws; (void)ws_size;
  const float* nf = (const float*)d_in[0];
  const int*   ei = (const int*)d_in[1];
  const float* W1 = (const float*)d_in[2];
  const float* b1 = (const float*)d_in[3];
  const float* W2 = (const float*)d_in[4];
  const float* b2 = (const float*)d_in[5];
  const float* gm = (const float*)d_in[6];
  const float* bt = (const float*)d_in[7];
  const float* Mw = (const float*)d_in[8];
  float* out = (float*)d_out;

  int n_edges = in_sizes[1] / 2;   // edge_index is [2, E]
  int n_pairs = n_edges / 32;      // 800000 -> 25000 pairs of 16-edge tiles

  dim3 grid(512), block(256);
  hipLaunchKernelGGL(qgnn_kernel, grid, block, LDS_TOTAL, stream,
                     nf, ei, W1, b1, W2, b2, gm, bt, Mw, out, n_edges, n_pairs);
}
